// FeatureTransfer_52372831207674
// MI455X (gfx1250) — matrix operand order repaired and checked by reading, not compiled, not run on hardware
//
#include <hip/hip_runtime.h>
#include <math.h>
#include <stdint.h>

typedef __attribute__((ext_vector_type(16))) _Float16 v16h;
typedef __attribute__((ext_vector_type(8)))  float    v8f;
typedef __attribute__((ext_vector_type(4)))  unsigned int v4u;
typedef __attribute__((ext_vector_type(8)))  unsigned int v8u;

#define KNN  16
#define CIN  64
#define COUT 128
#define TS   2048   // kNN source tile (3 * 2048 * 4B = 24 KB LDS)

__device__ __forceinline__ float silu_f(float x) {
    return x / (1.0f + __expf(-x));
}

// ---------------------------------------------------------------------------
// Kernel 1: convert weights f32 [K][N] row-major -> f16 transposed [N][K]
// so that the WMMA B-operand K-pairs (K=2v,2v+1) are contiguous in memory.
// ---------------------------------------------------------------------------
__global__ __launch_bounds__(256) void convert_weights_kernel(
    const float* __restrict__ Wlin, const float* __restrict__ Wsrc,
    const float* __restrict__ Wdst, const float* __restrict__ Wattn,
    _Float16* __restrict__ WlinT, _Float16* __restrict__ WsrcT,
    _Float16* __restrict__ WdstT, _Float16* __restrict__ WattnT)
{
    int t = blockIdx.x * blockDim.x + threadIdx.x;
    if (t < CIN * COUT) {
        int k = t / COUT, n = t % COUT;
        WlinT[n * CIN + k] = (_Float16)Wlin[t];
        WsrcT[n * CIN + k] = (_Float16)Wsrc[t];
        WdstT[n * CIN + k] = (_Float16)Wdst[t];
    }
    if (t < COUT * COUT) {
        int k = t / COUT, n = t % COUT;
        WattnT[n * COUT + k] = (_Float16)Wattn[t];
    }
}

// ---------------------------------------------------------------------------
// Kernel 2: brute-force kNN, one thread per target, LDS-tiled sources,
// register-resident sorted top-16 (fully unrolled insertion => no scratch).
// ---------------------------------------------------------------------------
__global__ __launch_bounds__(256) void knn_kernel(
    const float* __restrict__ spos, const float* __restrict__ tpos,
    int* __restrict__ knn_out, int Ns, int Nt)
{
    __shared__ float sx[TS], sy[TS], sz[TS];
    int t = blockIdx.x * 256 + threadIdx.x;
    bool valid = (t < Nt);
    float tx = 0.f, ty = 0.f, tz = 0.f;
    if (valid) { tx = tpos[3*t]; ty = tpos[3*t+1]; tz = tpos[3*t+2]; }

    float bd[KNN]; int bi[KNN];
#pragma unroll
    for (int q = 0; q < KNN; ++q) { bd[q] = 3.4e38f; bi[q] = 0; }

    for (int base = 0; base < Ns; base += TS) {
        int tile = min(TS, Ns - base);
        __syncthreads();
        for (int s = threadIdx.x; s < tile; s += 256) {
            sx[s] = spos[3*(base+s)+0];
            sy[s] = spos[3*(base+s)+1];
            sz[s] = spos[3*(base+s)+2];
        }
        // speculative prefetch of next tile -> global_prefetch_b8
        if (base + TS < Ns)
            __builtin_prefetch(&spos[3*(base+TS) + 3*(int)threadIdx.x], 0, 0);
        __syncthreads();
        if (valid) {
            for (int s = 0; s < tile; ++s) {
                float dx = tx - sx[s], dy = ty - sy[s], dz = tz - sz[s];
                float d = dx*dx + dy*dy + dz*dz;
                if (d < bd[KNN-1]) {
                    bd[KNN-1] = d; bi[KNN-1] = base + s;
#pragma unroll
                    for (int q = KNN-1; q > 0; --q) {
                        if (bd[q] < bd[q-1]) {
                            float td = bd[q]; bd[q] = bd[q-1]; bd[q-1] = td;
                            int   ti = bi[q]; bi[q] = bi[q-1]; bi[q-1] = ti;
                        }
                    }
                }
            }
        }
    }
    if (valid) {
#pragma unroll
        for (int q = 0; q < KNN; ++q) knn_out[t*KNN + q] = bi[q];
    }
}

// ---------------------------------------------------------------------------
// Kernel 3: Y[N,128] = X[N,64] @ W[64,128] via v_wmma_f32_16x16x32_f16.
// One wave per 16x16 output tile, 2 K-steps of 32.
// A layout (16-bit 16x32): lane l -> row M=l%16, VGPR v holds K-pair
//   ka = (v<4 ? 2v : 8+2v) + 8*(l/16).
// B layout (16-bit 32x16): lane l -> col N=l%16, VGPR v holds K-pair
//   kb = (v<4 ? 2v : 8+2v) + 8*(l/16)  (same as A).
// D layout: VGPR r -> row r+8g, col = n0 + l%16.
// ---------------------------------------------------------------------------
__global__ __launch_bounds__(128) void feat_gemm_kernel(
    const float* __restrict__ X, const _Float16* __restrict__ WT,
    float* __restrict__ Y, int Nrows)
{
    int wave = (blockIdx.x * blockDim.x + threadIdx.x) >> 5;
    int lane = threadIdx.x & 31;
    int m0 = (wave >> 3) * 16;          // COUT/16 == 8 n-tiles
    int n0 = (wave & 7) * 16;
    if (m0 >= Nrows) return;
    int m = lane & 15, g = lane >> 4;

    v8f acc = {};
#pragma unroll
    for (int ks = 0; ks < CIN / 32; ++ks) {
        v16h a, b;
        const float*    Ap = X  + (size_t)(m0 + m) * CIN + 32 * ks;
        const _Float16* Bp = WT + (size_t)(n0 + m) * CIN + 32 * ks;
#pragma unroll
        for (int v = 0; v < 8; ++v) {
            int ka = ((v < 4) ? 2*v : 8 + 2*v) + 8*g;
            a[2*v]   = (_Float16)Ap[ka];
            a[2*v+1] = (_Float16)Ap[ka+1];
            int kb = ((v < 4) ? 2*v : 8 + 2*v) + 8*g;
            b[2*v]   = Bp[kb];
            b[2*v+1] = Bp[kb+1];
        }
        acc = __builtin_amdgcn_wmma_f32_16x16x32_f16(
                  false, a, false, b, (short)0, acc, false, false);
    }
#pragma unroll
    for (int r = 0; r < 8; ++r)
        Y[(size_t)(m0 + r + 8*g) * COUT + (n0 + m)] = acc[r];
}

// ---------------------------------------------------------------------------
// Kernel 4: fused per-edge MLP. One 128-thread block (4 waves) per target,
// block loops over targets; W_attn staged once into LDS via the Tensor Data
// Mover (tensor_load_to_lds, TENSORcnt), then kept block-persistent.
//   A: delta = silu(rel@W_pos+b), alpha_pre = a_dst - a_src[j] + delta (->f16)
//   B: alpha = silu(alpha_pre[16x128] @ W_attn[128x128] + b)  [8 WMMA / wave]
//   C: per-channel softmax over k=16 + weighted gather-sum of val+delta.
// ---------------------------------------------------------------------------
__global__ __launch_bounds__(128) void edge_kernel(
    const float* __restrict__ spos, const float* __restrict__ tpos,
    const int* __restrict__ knn, const float* __restrict__ a_src,
    const float* __restrict__ a_dst, const float* __restrict__ val_src,
    const _Float16* __restrict__ WattnT, const float* __restrict__ Wpos,
    const float* __restrict__ bpos, const float* __restrict__ battn,
    float* __restrict__ out, int Nt)
{
    __shared__ _Float16 s_W[COUT * COUT];     // W_attn^T [n][k], 32 KB
    __shared__ _Float16 s_alpha[KNN][COUT];   // alpha_pre (WMMA A source)
    __shared__ float    s_delta[KNN][COUT];
    __shared__ float    s_att[KNN][COUT];     // silu(alpha)
    __shared__ float    s_adst[COUT];
    __shared__ float    s_wpos[3][COUT];
    __shared__ float    s_bpos[COUT], s_battn[COUT];
    __shared__ int      s_idx[KNN];
    __shared__ float    s_sp[KNN][3];
    __shared__ float    s_tp[3];

    int tid = threadIdx.x;

    // --- TDM: one-shot DMA of W_attn^T (8192 dwords) global -> LDS ---------
    // D# group0: count=1 | lds_addr | global_addr[56:0] | type=2
    // D# group1: data_size=4B, 1-D: tensor_dim0=tile_dim0=8192, dim1=1
    if (tid < 32) {                       // wave 0 only (TDM ignores EXEC)
        unsigned lds_off = (unsigned)(uintptr_t)&s_W[0];   // addr[31:0] = LDS offset
        unsigned long long ga = (unsigned long long)(uintptr_t)WattnT;
        v4u g0;
        g0[0] = 1u;                                        // count=1 (valid)
        g0[1] = lds_off;                                   // lds_addr
        g0[2] = (unsigned)(ga & 0xFFFFFFFFu);              // global_addr[31:0]
        g0[3] = (unsigned)((ga >> 32) & 0x1FFFFFFu)        // global_addr[56:32]
              | (2u << 30);                                // type = 2 (image)
        v8u g1;
        g1[0] = (2u << 16);          // workgroup_mask=0, data_size=2 (4B)
        g1[1] = (8192u << 16);       // tensor_dim0[15:0] << 16 (=8192)
        g1[2] = (1u << 16);          // tensor_dim0[31:16]=0, tensor_dim1=1
        g1[3] = (8192u << 16);       // tensor_dim1[31:16]=0, tile_dim0=8192
        g1[4] = 0u;                  // tile_dim1=0 (unused), tile_dim2=0
        g1[5] = 8192u;               // tensor_dim0_stride[31:0]
        g1[6] = 0u;                  // stride0[47:32]=0, stride1 lo=0
        g1[7] = 0u;                  // stride1 hi
        asm volatile("tensor_load_to_lds %0, %1" :: "s"(g0), "s"(g1) : "memory");
        __builtin_amdgcn_s_wait_tensorcnt(0);
    }
    // small per-block constants on the VALU path meanwhile
    s_wpos[0][tid] = Wpos[0*COUT + tid];
    s_wpos[1][tid] = Wpos[1*COUT + tid];
    s_wpos[2][tid] = Wpos[2*COUT + tid];
    s_bpos[tid]  = bpos[tid];
    s_battn[tid] = battn[tid];
    __syncthreads();

    int wave = tid >> 5, lane = tid & 31;
    int m = lane & 15, g = lane >> 4;

    for (int i = blockIdx.x; i < Nt; i += gridDim.x) {
        s_adst[tid] = a_dst[(size_t)i * COUT + tid];
        if (tid < KNN) {
            int j = knn[i * KNN + tid];
            s_idx[tid] = j;
            s_sp[tid][0] = spos[3*j]; s_sp[tid][1] = spos[3*j+1]; s_sp[tid][2] = spos[3*j+2];
        }
        if (tid < 3) s_tp[tid] = tpos[3*i + tid];
        __syncthreads();

        // Phase A: channel = tid, loop edges
        {
            int c = tid;
            float w0 = s_wpos[0][c], w1 = s_wpos[1][c], w2 = s_wpos[2][c];
            float bb = s_bpos[c], ad = s_adst[c];
            float txx = s_tp[0], tyy = s_tp[1], tzz = s_tp[2];
#pragma unroll
            for (int e = 0; e < KNN; ++e) {
                float rx = txx - s_sp[e][0], ry = tyy - s_sp[e][1], rz = tzz - s_sp[e][2];
                float dlt = silu_f(rx*w0 + ry*w1 + rz*w2 + bb);
                s_delta[e][c] = dlt;
                float ap = ad - a_src[(size_t)s_idx[e] * COUT + c] + dlt;
                s_alpha[e][c] = (_Float16)ap;
            }
        }
        __syncthreads();

        // Phase B: WMMA GEMM, each wave does two 16-col tiles, 4 K-steps each
#pragma unroll
        for (int nt = 0; nt < 2; ++nt) {
            int n0 = (wave * 2 + nt) * 16;
            v8f acc = {};
#pragma unroll
            for (int ks = 0; ks < COUT / 32; ++ks) {
                v16h a, b;
                const _Float16* Bp = &s_W[(n0 + m) * COUT + 32 * ks];
#pragma unroll
                for (int v = 0; v < 8; ++v) {
                    int ka = 32*ks + ((v < 4) ? 2*v : 8 + 2*v) + 8*g;
                    a[2*v]   = s_alpha[m][ka];
                    a[2*v+1] = s_alpha[m][ka + 1];
                    int kb = ((v < 4) ? 2*v : 8 + 2*v) + 8*g;
                    b[2*v]   = Bp[kb];
                    b[2*v+1] = Bp[kb + 1];
                }
                acc = __builtin_amdgcn_wmma_f32_16x16x32_f16(
                          false, a, false, b, (short)0, acc, false, false);
            }
            float bb = s_battn[n0 + m];
#pragma unroll
            for (int r = 0; r < 8; ++r)
                s_att[r + 8*g][n0 + m] = silu_f(acc[r] + bb);
        }
        __syncthreads();

        // Phase C: per-channel softmax over k + weighted gather-sum
        {
            int c = tid;
            float mx = -3.4e38f;
#pragma unroll
            for (int e = 0; e < KNN; ++e) mx = fmaxf(mx, s_att[e][c]);
            float w[KNN], sum = 0.0f;
#pragma unroll
            for (int e = 0; e < KNN; ++e) { w[e] = __expf(s_att[e][c] - mx); sum += w[e]; }
            float inv = 1.0f / sum, o = 0.0f;
#pragma unroll
            for (int e = 0; e < KNN; ++e) {
                float v = val_src[(size_t)s_idx[e] * COUT + c] + s_delta[e][c];
                o += (w[e] * inv) * v;
            }
            out[(size_t)i * COUT + c] = o;
        }
        __syncthreads();
    }
}

// ---------------------------------------------------------------------------
extern "C" void kernel_launch(void* const* d_in, const int* in_sizes, int n_in,
                              void* d_out, int out_size, void* d_ws, size_t ws_size,
                              hipStream_t stream)
{
    const float* spos  = (const float*)d_in[0];
    const float* tpos  = (const float*)d_in[1];
    const float* sfeat = (const float*)d_in[2];
    const float* tfeat = (const float*)d_in[3];
    const float* Wlin  = (const float*)d_in[4];
    const float* Wsrc  = (const float*)d_in[5];
    const float* Wdst  = (const float*)d_in[6];
    const float* Wpos  = (const float*)d_in[7];
    const float* bpos  = (const float*)d_in[8];
    const float* Wattn = (const float*)d_in[9];
    const float* battn = (const float*)d_in[10];
    float* out = (float*)d_out;

    int Ns = in_sizes[0] / 3;
    int Nt = in_sizes[1] / 3;

    char* ws = (char*)d_ws;
    size_t off = 0;
    int* knn = (int*)(ws + off);        off += (size_t)Nt * KNN * sizeof(int);
    off = (off + 255) & ~(size_t)255;
    float* val_src = (float*)(ws + off); off += (size_t)Ns * COUT * sizeof(float);
    float* a_src   = (float*)(ws + off); off += (size_t)Ns * COUT * sizeof(float);
    float* a_dst   = (float*)(ws + off); off += (size_t)Nt * COUT * sizeof(float);
    _Float16* WlinT  = (_Float16*)(ws + off); off += (size_t)CIN  * COUT * sizeof(_Float16);
    _Float16* WsrcT  = (_Float16*)(ws + off); off += (size_t)CIN  * COUT * sizeof(_Float16);
    _Float16* WdstT  = (_Float16*)(ws + off); off += (size_t)CIN  * COUT * sizeof(_Float16);
    _Float16* WattnT = (_Float16*)(ws + off); off += (size_t)COUT * COUT * sizeof(_Float16);

    convert_weights_kernel<<<(COUT*COUT + 255) / 256, 256, 0, stream>>>(
        Wlin, Wsrc, Wdst, Wattn, WlinT, WsrcT, WdstT, WattnT);

    knn_kernel<<<(Nt + 255) / 256, 256, 0, stream>>>(spos, tpos, knn, Ns, Nt);

    int wavesS  = ((Ns + 15) / 16) * (COUT / 16);
    int blocksS = (wavesS + 3) / 4;
    feat_gemm_kernel<<<blocksS, 128, 0, stream>>>(sfeat, WlinT, val_src, Ns);
    feat_gemm_kernel<<<blocksS, 128, 0, stream>>>(sfeat, WsrcT, a_src, Ns);

    int wavesT  = ((Nt + 15) / 16) * (COUT / 16);
    int blocksT = (wavesT + 3) / 4;
    feat_gemm_kernel<<<blocksT, 128, 0, stream>>>(tfeat, WdstT, a_dst, Nt);

    edge_kernel<<<1024, 128, 0, stream>>>(
        spos, tpos, knn, a_src, a_dst, val_src, WattnT, Wpos, bpos, battn, out, Nt);
}
